// Self_Attention_80994493268324
// MI455X (gfx1250) — matrix operand order repaired and checked by reading, not compiled, not run on hardware
//
#include <hip/hip_runtime.h>
#include <stdint.h>

// Causal self-attention, B=8,S=2048,E=1024,DK=DV=256, gfx1250 (MI455X).
// bf16 WMMA everywhere. Attention: K tiles via TDM (tensor_load_to_lds, TENSORcnt),
// V tiles via async global->LDS (ASYNCcnt), double buffered across a 4-wave block.

#define Bsz   8
#define Sseq  2048
#define Edim  1024
#define DKd   256
#define DVd   256
#define Mrows (Bsz * Sseq)
#define KBLK  32

typedef __attribute__((ext_vector_type(16))) __bf16 v16bf;
typedef __attribute__((ext_vector_type(8)))  float  v8f;
typedef unsigned int u32x4 __attribute__((ext_vector_type(4)));
typedef int          i32x8 __attribute__((ext_vector_type(8)));
typedef int          i32x4 __attribute__((ext_vector_type(4)));

union Frag {
    v16bf v;
    uint4 q[2];
};

__device__ __forceinline__ uint16_t f2bf(float f) {
    uint32_t u = __float_as_uint(f);
    u += 0x7FFFu + ((u >> 16) & 1u);   // round-to-nearest-even
    return (uint16_t)(u >> 16);
}

__device__ __forceinline__ v8f wmma_bf16(const Frag& a, const Frag& b, v8f c) {
    return __builtin_amdgcn_wmma_f32_16x16x32_bf16(false, a.v, false, b.v,
                                                   (short)0, c, false, false);
}

// TDM: move a 32-row x 512-byte K tile (rows of a [2048 x 512B] tensor) into LDS.
// D# per CDNA5 ISA 8.3/8.4: data_size=8B, dim0=64 units, stride0=64, dim1=2048,
// tile = 64 x 32, type=2, groups 2/3 zero (2-D tensor).
// This toolchain exposes the 6-arg builtin: (g0, g1, g2, g3, g4, cpol).
__device__ __forceinline__ void tdm_load_k(uint32_t lds_addr, const void* gptr) {
    uint64_t ga = (uint64_t)(uintptr_t)gptr;
    u32x4 g0;
    g0[0] = 1u;                                                // count=1 (valid user D#)
    g0[1] = lds_addr;                                          // lds_addr [63:32]
    g0[2] = (uint32_t)ga;                                      // global_addr[31:0]
    g0[3] = (uint32_t)((ga >> 32) & 0x01FFFFFFu) | (2u << 30); // gaddr[56:32] | type=2
    i32x8 g1;
    g1[0] = 0x00030000;            // wg_mask=0, data_size=3 (8B)
    g1[1] = (int)(64u << 16);      // tensor_dim0[15:0]=64
    g1[2] = (int)(2048u << 16);    // tensor_dim0[31:16]=0, tensor_dim1[15:0]=2048
    g1[3] = (int)(64u << 16);      // tensor_dim1[31:16]=0, tile_dim0=64
    g1[4] = 32;                    // tile_dim1=32, tile_dim2=0
    g1[5] = 64;                    // tensor_dim0_stride[31:0]=64
    g1[6] = 0;                     // stride0 hi=0, tensor_dim1_stride lo=0
    g1[7] = 0;
    i32x4 z4 = {0, 0, 0, 0};
    i32x8 z8 = {0, 0, 0, 0, 0, 0, 0, 0};
    __builtin_amdgcn_tensor_load_to_lds(g0, g1, z4, z4, z8, 0);
}

// ---------------- fp32 -> bf16 conversion of X ----------------
__global__ void cvt_x_kernel(const float* __restrict__ X,
                             uint16_t* __restrict__ Xb, int n4) {
    int i = blockIdx.x * blockDim.x + threadIdx.x;
    if (i >= n4) return;
    float4 f = ((const float4*)X)[i];
    union { uint16_t h[4]; uint64_t u; } o;
    o.h[0] = f2bf(f.x); o.h[1] = f2bf(f.y);
    o.h[2] = f2bf(f.z); o.h[3] = f2bf(f.w);
    ((uint64_t*)Xb)[i] = o.u;
}

// ------------- weight transpose + bf16: W[e][n] -> WT[n][e] -------------
__global__ void cvt_w_kernel(const float* __restrict__ Wq,
                             const float* __restrict__ Wk,
                             const float* __restrict__ Wv,
                             uint16_t* __restrict__ WqT,
                             uint16_t* __restrict__ WkT,
                             uint16_t* __restrict__ WvT) {
    int idx = blockIdx.x * blockDim.x + threadIdx.x;   // idx = e*DK + n
    if (idx >= Edim * DKd) return;
    int e = idx / DKd, n = idx % DKd;
    int o = n * Edim + e;
    WqT[o] = f2bf(Wq[idx]);
    WkT[o] = f2bf(Wk[idx]);
    WvT[o] = f2bf(Wv[idx]);
}

// ------------- projection GEMM: Out[M,DK] = Xb[M,E] * WT^T -------------
__global__ void proj_kernel(const uint16_t* __restrict__ Xb,
                            const uint16_t* __restrict__ WT,
                            uint16_t* __restrict__ Out) {
    const int lane = threadIdx.x & 31;
    const int lo = lane & 15, hi = lane >> 4;
    const int wid = blockIdx.x * (blockDim.x >> 5) + (threadIdx.x >> 5);
    const int ngrp = wid & 3;
    const int mt   = wid >> 2;
    const int m0 = mt * 32, n0 = ngrp * 64;

    v8f zero = {};
    v8f acc[2][4];
#pragma unroll
    for (int i = 0; i < 2; ++i)
#pragma unroll
        for (int t = 0; t < 4; ++t) acc[i][t] = zero;

    for (int c = 0; c < Edim / 32; ++c) {
        Frag a[2];
#pragma unroll
        for (int i = 0; i < 2; ++i) {
            const uint16_t* pa = Xb + (size_t)(m0 + i * 16 + lo) * Edim + c * 32 + hi * 8;
            a[i].q[0] = *(const uint4*)pa;
            a[i].q[1] = *(const uint4*)(pa + 16);
        }
#pragma unroll
        for (int t = 0; t < 4; ++t) {
            Frag bfr;
            const uint16_t* pb = WT + (size_t)(n0 + t * 16 + lo) * Edim + c * 32 + hi * 8;
            bfr.q[0] = *(const uint4*)pb;
            bfr.q[1] = *(const uint4*)(pb + 16);
#pragma unroll
            for (int i = 0; i < 2; ++i)
                acc[i][t] = wmma_bf16(a[i], bfr, acc[i][t]);
        }
    }
#pragma unroll
    for (int i = 0; i < 2; ++i)
#pragma unroll
        for (int t = 0; t < 4; ++t)
#pragma unroll
            for (int r = 0; r < 8; ++r)
                Out[(size_t)(m0 + i * 16 + r + 8 * hi) * DKd + n0 + t * 16 + lo] =
                    f2bf(acc[i][t][r]);
}

// ------- V-transposed GEMM: Vt[b][dv][s] = sum_e WvT[dv][e]*Xb[b*S+s][e] -------
__global__ void projv_kernel(const uint16_t* __restrict__ Xb,
                             const uint16_t* __restrict__ WvT,
                             uint16_t* __restrict__ Vt) {
    const int lane = threadIdx.x & 31;
    const int lo = lane & 15, hi = lane >> 4;
    const int wid = blockIdx.x * (blockDim.x >> 5) + (threadIdx.x >> 5);
    const int dvt = wid & 7;
    const int mg  = wid >> 3;
    const int dv0 = dvt * 32, m0 = mg * 64;

    v8f zero = {};
    v8f acc[2][4];
#pragma unroll
    for (int i = 0; i < 2; ++i)
#pragma unroll
        for (int t = 0; t < 4; ++t) acc[i][t] = zero;

    for (int c = 0; c < Edim / 32; ++c) {
        Frag a[2];
#pragma unroll
        for (int i = 0; i < 2; ++i) {
            const uint16_t* pa = WvT + (size_t)(dv0 + i * 16 + lo) * Edim + c * 32 + hi * 8;
            a[i].q[0] = *(const uint4*)pa;
            a[i].q[1] = *(const uint4*)(pa + 16);
        }
#pragma unroll
        for (int t = 0; t < 4; ++t) {
            Frag bfr;
            const uint16_t* pb = Xb + (size_t)(m0 + t * 16 + lo) * Edim + c * 32 + hi * 8;
            bfr.q[0] = *(const uint4*)pb;
            bfr.q[1] = *(const uint4*)(pb + 16);
#pragma unroll
            for (int i = 0; i < 2; ++i)
                acc[i][t] = wmma_bf16(a[i], bfr, acc[i][t]);
        }
    }
    const int b = m0 / Sseq;
    const int s_base = m0 % Sseq;
#pragma unroll
    for (int i = 0; i < 2; ++i)
#pragma unroll
        for (int t = 0; t < 4; ++t)
#pragma unroll
            for (int r = 0; r < 8; ++r)
                Vt[(size_t)(b * DVd + dv0 + i * 16 + r + 8 * hi) * Sseq + s_base + t * 16 + lo] =
                    f2bf(acc[i][t][r]);
}

// ---------------- causal flash attention ----------------
// 128-thread block = 4 waves = 4 consecutive 16-query tiles.
// K tile: TDM load (wave 0, TENSORcnt). V tile: async loads (all lanes, ASYNCcnt).
__global__ void __launch_bounds__(128) attn_kernel(const uint16_t* __restrict__ Qb,
                                                   const uint16_t* __restrict__ Kb,
                                                   const uint16_t* __restrict__ Vt,
                                                   float* __restrict__ Out) {
    __shared__ uint16_t Kbuf[2][KBLK][DKd];   // 2 x 16 KB, key-major
    __shared__ uint16_t Vbuf[2][DVd][KBLK];   // 2 x 16 KB, dv-major
    __shared__ uint16_t Pbuf[4][16 * 32];     // per-wave P staging

    const int tid  = threadIdx.x;
    const int lane = tid & 31;
    const int lo = lane & 15, hi = lane >> 4;
    const int w  = tid >> 5;                 // wave in block: 0..3
    const int b  = blockIdx.x >> 5;          // batch
    const int bq = blockIdx.x & 31;          // group of 4 query tiles
    const int qt = bq * 4 + w;
    const int q0 = qt * 16;

    const int my_nblk = qt / 2 + 1;          // causal 32-key blocks this wave needs
    const int nmax    = 2 * bq + 2;          // max over the 4 waves in this block

    const uint16_t* kg = Kb + (size_t)(b * Sseq) * DKd;
    const uint16_t* vg = Vt + (size_t)(b * DVd) * Sseq;

    const uint32_t ldsK0 = (uint32_t)(uintptr_t)&Kbuf[0][0][0];
    const uint32_t ldsV0 = (uint32_t)(uintptr_t)&Vbuf[0][0][0];

    auto issue_async = [](uint32_t lds, uint32_t goff, const void* base) {
        asm volatile("global_load_async_to_lds_b128 %0, %1, %2"
                     :: "v"(lds), "v"(goff), "s"(base) : "memory");
    };
    // V tile: 256 rows x 64 B -> 8 async b128 per thread
    auto issue_v = [&](int bufi, int k0) {
        uint32_t row0 = (uint32_t)(tid >> 2), col = (uint32_t)(tid & 3) * 16;
        uint32_t vl0 = ldsV0 + (uint32_t)bufi * (DVd * KBLK * 2) + row0 * (KBLK * 2) + col;
        uint32_t vg0 = row0 * (Sseq * 2) + (uint32_t)(k0 * 2) + col;
#pragma unroll
        for (int i = 0; i < 8; ++i)
            issue_async(vl0 + i * 32 * (KBLK * 2), vg0 + i * 32 * (Sseq * 2), vg);
    };
    auto issue_k = [&](int bufi, int k0) {     // callers guard with (w == 0)
        tdm_load_k(ldsK0 + (uint32_t)bufi * (KBLK * DKd * 2),
                   kg + (size_t)k0 * DKd);
    };

    // Q tile: 16 rows x 256 dk -> 8 A-fragments kept in registers
    Frag qf[8];
    {
        const uint16_t* pq = Qb + (size_t)(b * Sseq + q0 + lo) * DKd + hi * 8;
#pragma unroll
        for (int c = 0; c < 8; ++c) {
            qf[c].q[0] = *(const uint4*)(pq + c * 32);
            qf[c].q[1] = *(const uint4*)(pq + c * 32 + 16);
        }
    }

    v8f zero = {};
    v8f acc[16];
#pragma unroll
    for (int t = 0; t < 16; ++t) acc[t] = zero;
    float m_r[8], l_r[8];
#pragma unroll
    for (int r = 0; r < 8; ++r) { m_r[r] = -1e30f; l_r[r] = 0.0f; }

    if (w == 0) issue_k(0, 0);
    issue_v(0, 0);
    for (int blk = 0; blk < nmax; ++blk) {
        const int cur = blk & 1;
        if (blk + 1 < nmax) {
            if (w == 0) issue_k(cur ^ 1, (blk + 1) * KBLK);
            issue_v(cur ^ 1, (blk + 1) * KBLK);
            if (w == 0) __builtin_amdgcn_s_wait_tensorcnt(1);    // cur K tile landed
            asm volatile("s_wait_asynccnt 0x8" ::: "memory");    // cur V tile landed
        } else {
            if (w == 0) __builtin_amdgcn_s_wait_tensorcnt(0);
            asm volatile("s_wait_asynccnt 0x0" ::: "memory");
        }
        __syncthreads();                       // cur tiles visible to all waves

        if (blk < my_nblk) {                   // wave-uniform predicate: EXEC stays ~0
            const int k0 = blk * KBLK;
            v8f s0 = {}, s1 = {};
#pragma unroll
            for (int c = 0; c < 8; ++c) {
                Frag kf0, kf1;
                const uint16_t* pk0 = &Kbuf[cur][lo][c * 32 + hi * 8];
                kf0.q[0] = *(const uint4*)pk0;
                kf0.q[1] = *(const uint4*)(pk0 + 16);
                const uint16_t* pk1 = &Kbuf[cur][16 + lo][c * 32 + hi * 8];
                kf1.q[0] = *(const uint4*)pk1;
                kf1.q[1] = *(const uint4*)(pk1 + 16);
                s0 = wmma_bf16(qf[c], kf0, s0);
                s1 = wmma_bf16(qf[c], kf1, s1);
            }
            // online softmax; C-layout rows live per (r,hi); reduce over 16 lanes
            float alpha_r[8];
#pragma unroll
            for (int r = 0; r < 8; ++r) {
                const int qg = q0 + r + 8 * hi;
                float a0 = s0[r] * 0.0625f;    // 1/sqrt(256)
                float a1 = s1[r] * 0.0625f;
                if (k0 + lo > qg)      a0 = -1e30f;
                if (k0 + 16 + lo > qg) a1 = -1e30f;
                float mx = fmaxf(a0, a1);
                mx = fmaxf(mx, __shfl_xor(mx, 1));
                mx = fmaxf(mx, __shfl_xor(mx, 2));
                mx = fmaxf(mx, __shfl_xor(mx, 4));
                mx = fmaxf(mx, __shfl_xor(mx, 8));
                const float m_new = fmaxf(m_r[r], mx);
                const float alpha = __expf(m_r[r] - m_new);
                const float p0 = __expf(a0 - m_new);
                const float p1 = __expf(a1 - m_new);
                float ps = p0 + p1;
                ps += __shfl_xor(ps, 1);
                ps += __shfl_xor(ps, 2);
                ps += __shfl_xor(ps, 4);
                ps += __shfl_xor(ps, 8);
                l_r[r] = l_r[r] * alpha + ps;
                m_r[r] = m_new;
                alpha_r[r] = alpha;
                Pbuf[w][(r + 8 * hi) * 32 + lo]      = f2bf(p0);
                Pbuf[w][(r + 8 * hi) * 32 + 16 + lo] = f2bf(p1);
            }
#pragma unroll
            for (int t = 0; t < 16; ++t)
#pragma unroll
                for (int r = 0; r < 8; ++r)
                    acc[t][r] *= alpha_r[r];

            // wave-local cross-lane handoff via LDS (lockstep wave32)
            asm volatile("s_wait_dscnt 0x0" ::: "memory");
            Frag pf;
            {
                const uint16_t* pp = &Pbuf[w][lo * 32 + hi * 8];
                pf.q[0] = *(const uint4*)pp;
                pf.q[1] = *(const uint4*)(pp + 16);
            }
#pragma unroll
            for (int t = 0; t < 16; ++t) {
                Frag vf;
                const uint16_t* pv = &Vbuf[cur][t * 16 + lo][hi * 8];
                vf.q[0] = *(const uint4*)pv;
                vf.q[1] = *(const uint4*)(pv + 16);
                acc[t] = wmma_bf16(pf, vf, acc[t]);
            }
        }
        __syncthreads();                       // all readers done before buffer reuse
    }

    float inv_r[8];
#pragma unroll
    for (int r = 0; r < 8; ++r) inv_r[r] = 1.0f / l_r[r];
    float* outp = Out + (size_t)(b * Sseq + q0) * DVd;
#pragma unroll
    for (int t = 0; t < 16; ++t)
#pragma unroll
        for (int r = 0; r < 8; ++r)
            outp[(size_t)(r + 8 * hi) * DVd + t * 16 + lo] = acc[t][r] * inv_r[r];
}

// ---------------------------------------------------------------------------
extern "C" void kernel_launch(void* const* d_in, const int* in_sizes, int n_in,
                              void* d_out, int out_size, void* d_ws, size_t ws_size,
                              hipStream_t stream) {
    (void)in_sizes; (void)n_in; (void)out_size; (void)ws_size;
    const float* X  = (const float*)d_in[0];
    const float* Wq = (const float*)d_in[1];
    const float* Wk = (const float*)d_in[2];
    const float* Wv = (const float*)d_in[3];
    float* Out = (float*)d_out;
    char* ws = (char*)d_ws;

    // workspace layout (bytes), total ~60.3 MB
    uint16_t* Xb  = (uint16_t*)(ws);              // 33,554,432
    uint16_t* WqT = (uint16_t*)(ws + 33554432);   //    524,288
    uint16_t* WkT = (uint16_t*)(ws + 34078720);   //    524,288
    uint16_t* WvT = (uint16_t*)(ws + 34603008);   //    524,288
    uint16_t* Qb  = (uint16_t*)(ws + 35127296);   //  8,388,608
    uint16_t* Kb  = (uint16_t*)(ws + 43515904);   //  8,388,608
    uint16_t* Vt  = (uint16_t*)(ws + 51904512);   //  8,388,608

    const int n4 = Mrows * Edim / 4;
    cvt_x_kernel<<<(n4 + 255) / 256, 256, 0, stream>>>(X, Xb, n4);
    cvt_w_kernel<<<(Edim * DKd + 255) / 256, 256, 0, stream>>>(Wq, Wk, Wv, WqT, WkT, WvT);
    proj_kernel <<<256, 256, 0, stream>>>(Xb, WqT, Qb);   // Q
    proj_kernel <<<256, 256, 0, stream>>>(Xb, WkT, Kb);   // K
    projv_kernel<<<256, 256, 0, stream>>>(Xb, WvT, Vt);   // V^T
    attn_kernel <<<256, 128, 0, stream>>>(Qb, Kb, Vt, Out);
}
